// BidirectionalLSTM_54056458387944
// MI455X (gfx1250) — compile-verified, matrix operand order checked by reading, not run on hardware
//
#include <hip/hip_runtime.h>
#include <cstdint>
#include <cstddef>

#define T_SEQ 512
#define BATCH 64
#define D_IN  1024
#define UNITS 1024
#define KTOT  2048            // D_IN + UNITS
#define G4    4096            // 4 * UNITS
#define UB    32              // units per block
#define APITCH 40             // LDS row pitch in bf16 elems (80 B, 16B-aligned)
#define NCH   (KTOT / 32)     // 64 K-chunks

typedef __attribute__((ext_vector_type(16))) __bf16 bf16x16;
typedef __attribute__((ext_vector_type(8)))  float  f32x8;
typedef __attribute__((ext_vector_type(4)))  uint32_t u32x4;
typedef __attribute__((ext_vector_type(8)))  int      i32x8;
typedef __attribute__((ext_vector_type(4)))  int      i32x4;

union Frag { bf16x16 v; uint4 q[2]; };

#if __has_builtin(__builtin_amdgcn_tensor_load_to_lds)
#define HAVE_TDM 1
#else
#define HAVE_TDM 0
#endif

__device__ __forceinline__ unsigned short f2bf(float f) {
  union { float f; uint32_t u; } x{f};
  uint32_t u = x.u;
  uint32_t r = u + 0x7FFFu + ((u >> 16) & 1u);
  return (unsigned short)(r >> 16);
}

__device__ __forceinline__ float sigmoidf(float v) {
  return 1.0f / (1.0f + __expf(-v));
}

// Per-lane async 16B copy global -> LDS (ASYNCcnt). VDST carries the LDS byte
// address (low 32 bits of the generic pointer == wave LDS offset per aperture rules).
__device__ __forceinline__ void async_cp16(const void* g, void* lds) {
  uint32_t off = (uint32_t)(uintptr_t)lds;
  asm volatile("global_load_async_to_lds_b128 %0, %1, off"
               :: "v"(off), "v"(g) : "memory");
}

__device__ __forceinline__ void wait_async0() {
#if __has_builtin(__builtin_amdgcn_s_wait_asynccnt)
  __builtin_amdgcn_s_wait_asynccnt(0);
#else
  asm volatile("s_wait_asynccnt 0" ::: "memory");
#endif
}

#if HAVE_TDM
// 2D TDM tile load: bf16 elements, tile (tile_k x tile_rows), global row stride
// row_stride elems, LDS padded 4 DWORDs (16B) after every 16 DWORDs (64B) so a
// 32-elem row lands on an 80B pitch (== APITCH).
// Toolchain carries the 6-arg builtin form:
//   (uint32x4 g0, int32x8 g1, int32x4 g2, int32x4 g3, int32x8 g4, i32 cpol)
__device__ __forceinline__ void tdm_load_2d(void* lds, const void* gaddr,
                                            uint32_t tile_k, uint32_t tile_rows,
                                            uint32_t tensor_k, uint32_t tensor_rows,
                                            uint32_t row_stride) {
  uint64_t ga = (uint64_t)(uintptr_t)gaddr;
  u32x4 g0;
  g0[0] = 1u;                                   // count=1 (valid), no gather
  g0[1] = (uint32_t)(uintptr_t)lds;             // lds_addr (bytes)
  g0[2] = (uint32_t)(ga & 0xFFFFFFFFu);         // global_addr[31:0]
  g0[3] = (uint32_t)((ga >> 32) & 0x01FFFFFFu)  // global_addr[56:32]
        | (2u << 30);                           // type=2 ("image")
  i32x8 g1;
  g1[0] = (int)((1u << 16)        // data_size = 1 -> 2 bytes
        | (1u << 20)              // pad_enable
        | (3u << 22)              // pad_interval code 3 -> every 16 DWORDs (64B)
        | (3u << 25));            // pad_amount   code 3 -> 4 DWORDs (16B)
  g1[1] = (int)((tensor_k & 0xFFFFu) << 16);                            // tensor_dim0[15:0]
  g1[2] = (int)(((tensor_k >> 16) & 0xFFFFu) | ((tensor_rows & 0xFFFFu) << 16)); // dim0 hi | dim1 lo
  g1[3] = (int)(((tensor_rows >> 16) & 0xFFFFu) | ((tile_k & 0xFFFFu) << 16));   // dim1 hi | tile_dim0
  g1[4] = (int)(tile_rows & 0xFFFFu);                                   // tile_dim1 (tile_dim2=0)
  g1[5] = (int)row_stride;                                              // tensor_dim0_stride lo
  g1[6] = 0;
  g1[7] = 0;
  i32x4 z4 = {0, 0, 0, 0};
  i32x8 z8 = {0, 0, 0, 0, 0, 0, 0, 0};
  __builtin_amdgcn_tensor_load_to_lds(g0, g1, z4, z4, z8, 0);
}
#endif

__device__ __forceinline__ void wait_tensor0() {
#if __has_builtin(__builtin_amdgcn_s_wait_tensorcnt)
  __builtin_amdgcn_s_wait_tensorcnt(0);
#else
  asm volatile("s_wait_tensorcnt 0" ::: "memory");
#endif
}

// ---------------------------------------------------------------------------
// One-time prep: W [2048,4096] fp32 -> W^T [4096,2048] bf16 (both directions).
// ---------------------------------------------------------------------------
__global__ void wtrans_kernel(const float* __restrict__ Wf,
                              const float* __restrict__ Wb,
                              unsigned short* __restrict__ WbfT) {
  __shared__ float tile[32][33];
  const int d = blockIdx.z;
  const float* W = d ? Wb : Wf;
  unsigned short* dst = WbfT + (size_t)d * G4 * KTOT;
  const int k0 = blockIdx.x * 32;
  const int n0 = blockIdx.y * 32;
  const int tx = threadIdx.x;   // 0..31
  const int ty = threadIdx.y;   // 0..7
#pragma unroll
  for (int r = 0; r < 4; ++r) {
    int kk = ty * 4 + r;
    tile[kk][tx] = W[(size_t)(k0 + kk) * G4 + n0 + tx];
  }
  __syncthreads();
#pragma unroll
  for (int r = 0; r < 4; ++r) {
    int nn = ty * 4 + r;
    dst[(size_t)(n0 + nn) * KTOT + k0 + tx] = f2bf(tile[tx][nn]);
  }
}

// One-time prep: x fp32 -> bf16 (same [B,T,D] layout); 8 elements per thread.
__global__ void xconv_kernel(const float* __restrict__ x,
                             unsigned short* __restrict__ xbf) {
  size_t i = ((size_t)blockIdx.x * blockDim.x + threadIdx.x) * 8;
  if (i >= (size_t)BATCH * T_SEQ * D_IN) return;
  float4 f0 = *(const float4*)(x + i);
  float4 f1 = *(const float4*)(x + i + 4);
  uint32_t p0 = f2bf(f0.x) | ((uint32_t)f2bf(f0.y) << 16);
  uint32_t p1 = f2bf(f0.z) | ((uint32_t)f2bf(f0.w) << 16);
  uint32_t p2 = f2bf(f1.x) | ((uint32_t)f2bf(f1.y) << 16);
  uint32_t p3 = f2bf(f1.z) | ((uint32_t)f2bf(f1.w) << 16);
  *(uint4*)(xbf + i) = make_uint4(p0, p1, p2, p3);
}

__global__ void init_kernel(unsigned short* __restrict__ hbf,
                            float* __restrict__ cst) {
  int i = blockIdx.x * blockDim.x + threadIdx.x;
  if (i < 2 * BATCH * UNITS) { hbf[i] = 0; cst[i] = 0.0f; }
}

// ---------------------------------------------------------------------------
// One LSTM time-step, both directions (blockIdx.y = dir). 8 wave32s compute
// batch(64) x [4 gates x 32 units]; K = 2048 in double-buffered 32-wide chunks:
// A via async global->LDS copies (ASYNCcnt), B via TDM 2D tile loads (TENSORcnt),
// math via v_wmma_f32_16x16x32_bf16.
// ---------------------------------------------------------------------------
__global__ __launch_bounds__(256) void lstm_step_kernel(
    const unsigned short* __restrict__ xbf,   // [64, 512, 1024] bf16
    const unsigned short* __restrict__ WbfT,  // [2, 4096, 2048] bf16 (W^T)
    const float* __restrict__ bias_f,
    const float* __restrict__ bias_b,
    unsigned short* __restrict__ hbf,         // [2, 64, 1024] bf16 state
    float* __restrict__ cst,                  // [2, 64, 1024] fp32 state
    float* __restrict__ out,                  // [64, 512, 2048] fp32
    int t)
{
  __shared__ __align__(16) unsigned short sA[2][BATCH * APITCH]; // 64x32 bf16, dbl buf
  __shared__ __align__(16) unsigned short sB[2][128 * APITCH];   // 128x32 bf16, dbl buf
  __shared__ float sZ[BATCH][4 * UB];

  const int d     = blockIdx.y;
  const int u0    = blockIdx.x * UB;
  const int t_eff = d ? (T_SEQ - 1 - t) : t;
  const unsigned short* Wd = WbfT + (size_t)d * G4 * KTOT;
  unsigned short* hd = hbf + d * BATCH * UNITS;
  float*          cd = cst + d * BATCH * UNITS;
  const float*  bias = d ? bias_b : bias_f;

  const int tid    = threadIdx.x;
  const int lane   = tid & 31;
  const int wave   = tid >> 5;
  const int lane16 = lane & 15;
  const int half   = lane >> 4;
  const int kb     = half * 8;
  const int mblk   = wave & 3;
  const int nhalf  = wave >> 2;

  f32x8 acc[4] = {};

  // A staging role: each thread async-copies one 16B (8 bf16) piece per chunk
  const int ar  = tid >> 2;            // row 0..63 (batch)
  const int ac0 = (tid & 3) * 8;       // k sub-offset
  const unsigned short* xrow = xbf + (size_t)ar * (T_SEQ * D_IN) + (size_t)t_eff * D_IN + ac0;
  const unsigned short* hrow = hd + ar * UNITS + ac0;
#if !HAVE_TDM
  const int bcol = tid >> 1;
  const int bk   = (tid & 1) * 16;
  const unsigned short* bsrc_base =
      Wd + (size_t)((bcol >> 5) * UNITS + u0 + (bcol & 31)) * KTOT + bk;
#endif

#define STAGE(bufi, ch)                                                          \
  {                                                                              \
    const int k0_ = (ch) * 32;                                                   \
    const unsigned short* asrc = (k0_ < D_IN) ? (xrow + k0_)                     \
                                              : (hrow + (k0_ - D_IN));           \
    async_cp16(asrc, &sA[bufi][ar * APITCH + ac0]);                              \
    if (wave == 0) {                                                             \
      _Pragma("unroll")                                                          \
      for (int g_ = 0; g_ < 4; ++g_) {                                           \
        TDM_OR_ASYNC_B(bufi, g_, k0_);                                           \
      }                                                                          \
    }                                                                            \
    TDM_FALLBACK_B(bufi, k0_);                                                   \
  }

#if HAVE_TDM
#define TDM_OR_ASYNC_B(bufi, g_, k0_)                                            \
  tdm_load_2d(&sB[bufi][(g_)*32*APITCH],                                         \
              Wd + (size_t)((g_)*UNITS + u0) * KTOT + (k0_),                     \
              32u, 32u, (uint32_t)KTOT, (uint32_t)G4, (uint32_t)KTOT)
#define TDM_FALLBACK_B(bufi, k0_)
#else
#define TDM_OR_ASYNC_B(bufi, g_, k0_)
#define TDM_FALLBACK_B(bufi, k0_)                                                \
  {                                                                              \
    const unsigned short* bs = bsrc_base + (k0_);                                \
    async_cp16(bs,     &sB[bufi][bcol * APITCH + bk]);                           \
    async_cp16(bs + 8, &sB[bufi][bcol * APITCH + bk + 8]);                       \
  }
#endif

  // prologue: stage chunk 0 into buffer 0
  STAGE(0, 0);
  wait_async0();
  if (wave == 0) wait_tensor0();
  __syncthreads();

  for (int chunk = 0; chunk < NCH; ++chunk) {
    const int buf = chunk & 1;
    if (chunk + 1 < NCH) {
      STAGE(buf ^ 1, chunk + 1);   // async/TDM fill of next buffer overlaps WMMA
    }

    Frag a;
    const unsigned short* arow = &sA[buf][(mblk * 16 + lane16) * APITCH + kb];
    a.q[0] = *(const uint4*)(arow);
    a.q[1] = *(const uint4*)(arow + 16);
#pragma unroll
    for (int nt = 0; nt < 4; ++nt) {
      Frag b;
      const unsigned short* brow = &sB[buf][(nhalf * 64 + nt * 16 + lane16) * APITCH + kb];
      b.q[0] = *(const uint4*)(brow);
      b.q[1] = *(const uint4*)(brow + 16);
      acc[nt] = __builtin_amdgcn_wmma_f32_16x16x32_bf16(
          false, a.v, false, b.v, (short)0, acc[nt], false, false);
    }

    wait_async0();
    if (wave == 0) wait_tensor0();
    __syncthreads();
  }

  // scatter accumulators to LDS (C layout: VGPR r -> M = r + 8*half)
#pragma unroll
  for (int nt = 0; nt < 4; ++nt) {
    int col = nhalf * 64 + nt * 16 + lane16;
#pragma unroll
    for (int r = 0; r < 8; ++r) {
      int row = mblk * 16 + half * 8 + r;
      sZ[row][col] = acc[nt][r];
    }
  }
  __syncthreads();

  // LSTM cell update: each thread owns 8 units of one batch row
  const int eb  = tid >> 2;
  const int eu0 = (tid & 3) * 8;
#pragma unroll
  for (int j = 0; j < 8; ++j) {
    int uu = eu0 + j;
    int ug = u0 + uu;
    float zi = sZ[eb][0 * UB + uu] + bias[0 * UNITS + ug];
    float zf = sZ[eb][1 * UB + uu] + bias[1 * UNITS + ug];
    float zo = sZ[eb][2 * UB + uu] + bias[2 * UNITS + ug];
    float zg = sZ[eb][3 * UB + uu] + bias[3 * UNITS + ug];
    float cold = cd[eb * UNITS + ug];
    float cnew = sigmoidf(zf) * cold + sigmoidf(zi) * tanhf(zg);
    float hnew = sigmoidf(zo) * tanhf(cnew);
    cd[eb * UNITS + ug] = cnew;
    hd[eb * UNITS + ug] = f2bf(hnew);
    out[(size_t)eb * (T_SEQ * 2 * UNITS) + (size_t)t_eff * (2 * UNITS) + d * UNITS + ug] = hnew;
  }
}

// ---------------------------------------------------------------------------
extern "C" void kernel_launch(void* const* d_in, const int* in_sizes, int n_in,
                              void* d_out, int out_size, void* d_ws, size_t ws_size,
                              hipStream_t stream) {
  const float* x  = (const float*)d_in[0];
  const float* Wf = (const float*)d_in[1];
  const float* bf = (const float*)d_in[2];
  const float* Wb = (const float*)d_in[3];
  const float* bb = (const float*)d_in[4];
  float* out = (float*)d_out;

  char* ws = (char*)d_ws;
  const size_t wbytes = (size_t)2 * G4 * KTOT * sizeof(unsigned short);        // 32 MB
  const size_t xbytes = (size_t)BATCH * T_SEQ * D_IN * sizeof(unsigned short); // 64 MB
  const size_t hbytes = (size_t)2 * BATCH * UNITS * sizeof(unsigned short);
  unsigned short* WbfT = (unsigned short*)ws;
  unsigned short* xbf  = (unsigned short*)(ws + wbytes);
  unsigned short* hbf  = (unsigned short*)(ws + wbytes + xbytes);
  float*          cst  = (float*)(ws + wbytes + xbytes + hbytes);

  wtrans_kernel<<<dim3(KTOT / 32, G4 / 32, 2), dim3(32, 8), 0, stream>>>(Wf, Wb, WbfT);
  xconv_kernel<<<((size_t)BATCH * T_SEQ * D_IN / 8 + 255) / 256, 256, 0, stream>>>(x, xbf);
  init_kernel<<<(2 * BATCH * UNITS + 255) / 256, 256, 0, stream>>>(hbf, cst);
  for (int t = 0; t < T_SEQ; ++t) {
    lstm_step_kernel<<<dim3(UNITS / UB, 2), 256, 0, stream>>>(
        xbf, WbfT, bf, bb, hbf, cst, out, t);
  }
}
